// MultiHeadAttentionClassical_65481071409503
// MI455X (gfx1250) — matrix operand order repaired and checked by reading, not compiled, not run on hardware
//
#include <hip/hip_runtime.h>

typedef __attribute__((ext_vector_type(16))) _Float16 v16h;
typedef __attribute__((ext_vector_type(8)))  _Float16 v8h;
typedef __attribute__((ext_vector_type(8)))  float    v8f;
typedef __attribute__((ext_vector_type(4))) unsigned int u32x4;
typedef __attribute__((ext_vector_type(8))) int          i32x8;
typedef __attribute__((ext_vector_type(4))) int          i32x4;

#define WMMA_F16(a, b, c) \
  __builtin_amdgcn_wmma_f32_16x16x32_f16(false, (a), false, (b), (short)0, (c), false, false)

union V16 { v16h v; v8h h[2]; };

// ---------------------------------------------------------------------------
// v_permlane16_b32 xor-butterfly reductions within each 16-lane row (VALU,
// no DS/wait). Lane-select tables: nibble i selects lane (i ^ mask).
// ---------------------------------------------------------------------------
template <unsigned LO, unsigned HI>
__device__ __forceinline__ float permlane16_f(float v) {
  unsigned u = __builtin_bit_cast(unsigned, v);
  unsigned r = __builtin_amdgcn_permlane16(u, u, LO, HI, false, false);
  return __builtin_bit_cast(float, r);
}
__device__ __forceinline__ float row16_max(float v) {
  v = fmaxf(v, permlane16_f<0x67452301u, 0xEFCDAB89u>(v));  // xor 1
  v = fmaxf(v, permlane16_f<0x54761032u, 0xDCFE98BAu>(v));  // xor 2
  v = fmaxf(v, permlane16_f<0x32107654u, 0xBA98FEDCu>(v));  // xor 4
  v = fmaxf(v, permlane16_f<0xFEDCBA98u, 0x76543210u>(v));  // xor 8
  return v;
}
__device__ __forceinline__ float row16_sum(float v) {
  v += permlane16_f<0x67452301u, 0xEFCDAB89u>(v);
  v += permlane16_f<0x54761032u, 0xDCFE98BAu>(v);
  v += permlane16_f<0x32107654u, 0xBA98FEDCu>(v);
  v += permlane16_f<0xFEDCBA98u, 0x76543210u>(v);
  return v;
}

// ---------------------------------------------------------------------------
// TDM descriptor (D# groups 0/1 per CDNA5 ISA) for a 2D
// tile (tile_w x tile_h) of a row-major tensor [tensor_h][tensor_w] with row
// stride stride_elems. ds_code: 0=1B,1=2B,2=4B elements.
// ---------------------------------------------------------------------------
__device__ __forceinline__ void tdm_desc(u32x4& g0, i32x8& g1, unsigned lds_off,
                                         const void* gaddr, unsigned ds_code,
                                         unsigned tensor_w, unsigned tensor_h,
                                         unsigned stride_elems,
                                         unsigned tile_w, unsigned tile_h) {
  unsigned long long ga = (unsigned long long)(uintptr_t)gaddr;
  g0[0] = 1u;                                                // count=1, user mode
  g0[1] = lds_off;                                           // lds_addr (bytes)
  g0[2] = (unsigned)(ga & 0xFFFFFFFFu);                      // global_addr[31:0]
  g0[3] = (unsigned)((ga >> 32) & 0x01FFFFFFu) | (2u << 30); // addr[56:32]|type=2
  g1[0] = (int)(ds_code << 16);                              // data_size
  g1[1] = (int)((tensor_w & 0xFFFFu) << 16);                 // tensor_dim0 lo
  g1[2] = (int)(((tensor_w >> 16) & 0xFFFFu) | ((tensor_h & 0xFFFFu) << 16));
  g1[3] = (int)(((tensor_h >> 16) & 0xFFFFu) | ((tile_w & 0xFFFFu) << 16));
  g1[4] = (int)(tile_h & 0xFFFFu);                           // tile_dim1
  g1[5] = (int)stride_elems;                                 // dim0_stride lo
  g1[6] = 0;
  g1[7] = 0;
}
__device__ __forceinline__ void tdm_load(unsigned lds_off, const void* gaddr,
                                         unsigned ds_code, unsigned tw, unsigned th,
                                         unsigned stride, unsigned tile_w, unsigned tile_h) {
  u32x4 g0; i32x8 g1; i32x4 z = {};
  tdm_desc(g0, g1, lds_off, gaddr, ds_code, tw, th, stride, tile_w, tile_h);
#if __clang_major__ >= 23
  i32x8 z8 = {};
  __builtin_amdgcn_tensor_load_to_lds(g0, g1, z, z, z8, 0);
#else
  __builtin_amdgcn_tensor_load_to_lds(g0, g1, z, z, 0);
#endif
}
__device__ __forceinline__ void tdm_store(unsigned lds_off, void* gaddr,
                                          unsigned ds_code, unsigned tw, unsigned th,
                                          unsigned stride, unsigned tile_w, unsigned tile_h) {
  u32x4 g0; i32x8 g1; i32x4 z = {};
  tdm_desc(g0, g1, lds_off, gaddr, ds_code, tw, th, stride, tile_w, tile_h);
#if __clang_major__ >= 23
  i32x8 z8 = {};
  __builtin_amdgcn_tensor_store_from_lds(g0, g1, z, z, z8, 0);
#else
  __builtin_amdgcn_tensor_store_from_lds(g0, g1, z, z, 0);
#endif
}

// ---------------------------------------------------------------------------
// fp32 -> f16 elementwise conversion (grid-stride)
// ---------------------------------------------------------------------------
__global__ void cvt_f32_f16_kernel(const float* __restrict__ src,
                                   _Float16* __restrict__ dst, int n) {
  int i = blockIdx.x * blockDim.x + threadIdx.x;
  int stride = gridDim.x * blockDim.x;
  for (; i < n; i += stride) dst[i] = (_Float16)src[i];
}

// ---------------------------------------------------------------------------
// GEMM: out[M=8192][N=1024] = A[M][1024] * W[N][1024]^T  (y = x @ W^T)
// Block = 256 threads = 8 waves; tile 128 rows x 64 cols; K-step 32.
// A/W tiles double-buffered in LDS via TDM; output staged in LDS and written
// with one tensor_store per block.
// MODE 0: Q -> [bh][s][64]   MODE 1: K -> [bh][s][64]
// MODE 2: V -> [bh][64][s] (transposed)   MODE 3: O -> fp32 [M][1024]
// ---------------------------------------------------------------------------
template <int MODE>
__global__ __launch_bounds__(256, 2)
void proj_kernel(const _Float16* __restrict__ A,
                 const _Float16* __restrict__ W,
                 _Float16* __restrict__ out16,
                 float* __restrict__ out32) {
  __shared__ __align__(64) _Float16 ldsA[2][128 * 32];
  __shared__ __align__(64) _Float16 ldsW[2][64 * 32];
  __shared__ __align__(64) unsigned char ldsOut[(MODE == 3) ? (128 * 64 * 4)
                                                            : (128 * 64 * 2)];
  const int lane = threadIdx.x & 31;
  const int wave = threadIdx.x >> 5;
  const int half = lane >> 4;
  const int ln16 = lane & 15;
  const int block_m = blockIdx.y * 128;
  const int nbase = blockIdx.x * 64;
  const int mloc = wave * 16 + ln16;

  const _Float16* Ab = A + (size_t)block_m * 1024;        // 128 x 1024 slab
  const _Float16* Wb = W + (size_t)nbase * 1024;          // 64 x 1024 slab

  if (wave == 0) {  // prime buffer 0 (k0 = 0)
    tdm_load((unsigned)(uintptr_t)&ldsA[0][0], Ab, 1, 1024, 8192, 1024, 32, 128);
    tdm_load((unsigned)(uintptr_t)&ldsW[0][0], Wb, 1, 1024, 1024, 1024, 32, 64);
  }

  v8f c[4];
#pragma unroll
  for (int t = 0; t < 4; ++t) c[t] = (v8f){};

  for (int it = 0; it < 32; ++it) {
    if (wave == 0) {
      if (it + 1 < 32) {
        const int nb = (it + 1) & 1;
        const int k1 = (it + 1) * 32;
        tdm_load((unsigned)(uintptr_t)&ldsA[nb][0], Ab + k1, 1, 1024, 8192, 1024, 32, 128);
        tdm_load((unsigned)(uintptr_t)&ldsW[nb][0], Wb + k1, 1, 1024, 1024, 1024, 32, 64);
        __builtin_amdgcn_s_wait_tensorcnt(2);  // oldest pair (current buf) done
      } else {
        __builtin_amdgcn_s_wait_tensorcnt(0);
      }
    }
    __syncthreads();
    const _Float16* At = ldsA[it & 1];  // [128][32]
    const _Float16* Wt = ldsW[it & 1];  // [64][32]

    V16 a;
    const _Float16* arow = At + mloc * 32;
    a.h[0] = *(const v8h*)(arow + half * 16);
    a.h[1] = *(const v8h*)(arow + half * 16 + 8);
#pragma unroll
    for (int t = 0; t < 4; ++t) {
      v16h b = *(const v16h*)(Wt + (t * 16 + ln16) * 32 + half * 16);
      c[t] = WMMA_F16(a.v, b, c[t]);
    }
    __syncthreads();
  }

  // ---- stage output tile in LDS, then one TDM store per block ----
#pragma unroll
  for (int t = 0; t < 4; ++t) {
#pragma unroll
    for (int r = 0; r < 8; ++r) {
      const int ml = wave * 16 + r + 8 * half;  // local row 0..127
      const int nl = t * 16 + ln16;             // local col 0..63
      if (MODE == 3) ((float*)ldsOut)[ml * 64 + nl] = c[t][r];
      else if (MODE == 2) ((_Float16*)ldsOut)[nl * 128 + ml] = (_Float16)c[t][r];
      else ((_Float16*)ldsOut)[ml * 64 + nl] = (_Float16)c[t][r];
    }
  }
  __syncthreads();
  if (wave == 0) {
    const unsigned lo = (unsigned)(uintptr_t)&ldsOut[0];
    const int b = block_m >> 11, s0 = block_m & 2047, h = nbase >> 6;
    if (MODE == 3) {
      tdm_store(lo, out32 + (size_t)block_m * 1024 + nbase, 2,
                1024, 8192, 1024, 64, 128);
    } else if (MODE == 2) {  // V^T: [bh][64 d][2048 s]
      tdm_store(lo, out16 + ((size_t)(b * 16 + h) * 64) * 2048 + s0, 1,
                2048, 64, 2048, 128, 64);
    } else {                 // Q/K: [bh][2048 s][64 d]
      tdm_store(lo, out16 + ((size_t)(b * 16 + h) * 2048 + s0) * 64, 1,
                64, 2048, 64, 64, 128);
    }
    __builtin_amdgcn_s_wait_tensorcnt(0);  // keep LDS alive until DMA done
  }
}

// ---------------------------------------------------------------------------
// Flash attention: 1 block = 8 waves, each wave owns 16 query rows (128/block).
// grid = (64 bh, 16 qtiles). K/V tiles double-buffered in LDS via TDM;
// output tile staged in LDS and stored with one tensor_store.
// ---------------------------------------------------------------------------
__global__ __launch_bounds__(256, 2)
void attn_kernel(const _Float16* __restrict__ Q,   // [64][2048][64]
                 const _Float16* __restrict__ K,   // [64][2048][64]
                 const _Float16* __restrict__ Vt,  // [64][64][2048]
                 _Float16* __restrict__ AO) {      // [4][2048][1024]
  __shared__ __align__(64) _Float16 ldsK[2][64 * 64];  // [key][d]
  __shared__ __align__(64) _Float16 ldsV[2][64 * 64];  // [d][key]
  __shared__ __align__(64) _Float16 pbuf[8][16 * 64];  // P staging / out tile

  const int lane = threadIdx.x & 31;
  const int wave = threadIdx.x >> 5;
  const int half = lane >> 4;
  const int ln16 = lane & 15;
  const int bh = blockIdx.x;               // 0..63
  const int b = bh >> 4, h = bh & 15;
  const int q0 = blockIdx.y * 128;
  const int qbase = q0 + wave * 16;

  const _Float16* Qh = Q + (size_t)bh * 2048 * 64;
  const _Float16* Kh = K + (size_t)bh * 2048 * 64;
  const _Float16* Vh = Vt + (size_t)bh * 64 * 2048;
  _Float16* pw = pbuf[wave];

  if (wave == 0) {  // prime buffer 0 with the first K/V tile
    tdm_load((unsigned)(uintptr_t)&ldsK[0][0], Kh, 1, 64, 2048, 64, 64, 64);
    tdm_load((unsigned)(uintptr_t)&ldsV[0][0], Vh, 1, 2048, 64, 2048, 64, 64);
  }

  // Q A-fragments (d = 0..63 -> two K-steps of 32), resident in registers.
  V16 aq[2];
  {
    const _Float16* qrow = Qh + (size_t)(qbase + ln16) * 64;
#pragma unroll
    for (int ks = 0; ks < 2; ++ks) {
      aq[ks].h[0] = *(const v8h*)(qrow + ks * 32 + half * 16);
      aq[ks].h[1] = *(const v8h*)(qrow + ks * 32 + half * 16 + 8);
    }
  }

  float mrow[8], lrow[8];
  v8f o[4];
#pragma unroll
  for (int r = 0; r < 8; ++r) { mrow[r] = -1e30f; lrow[r] = 0.f; }
#pragma unroll
  for (int t = 0; t < 4; ++t) o[t] = (v8f){};

  const float scale = 0.125f;  // 1/sqrt(64)

  for (int it = 0; it < 32; ++it) {
    const int kb = it * 64;
    if (wave == 0) {
      if (it + 1 < 32) {
        const int nb = (it + 1) & 1;
        tdm_load((unsigned)(uintptr_t)&ldsK[nb][0],
                 Kh + (size_t)(kb + 64) * 64, 1, 64, 2048, 64, 64, 64);
        tdm_load((unsigned)(uintptr_t)&ldsV[nb][0],
                 Vh + (kb + 64), 1, 2048, 64, 2048, 64, 64);
        __builtin_amdgcn_s_wait_tensorcnt(2);
      } else {
        __builtin_amdgcn_s_wait_tensorcnt(0);
      }
    }
    __syncthreads();
    const _Float16* Kt = ldsK[it & 1];
    const _Float16* Vb = ldsV[it & 1];

    // ---- scores: S = Q K^T for 16 q-rows x 64 keys (B frags from LDS) ----
    v8f s[4];
#pragma unroll
    for (int t = 0; t < 4; ++t) {
      const _Float16* krow = Kt + (t * 16 + ln16) * 64;
      v8f acc = (v8f){};
      v16h bk0 = *(const v16h*)(krow + 0 + half * 16);
      v16h bk1 = *(const v16h*)(krow + 32 + half * 16);
      acc = WMMA_F16(aq[0].v, bk0, acc);
      acc = WMMA_F16(aq[1].v, bk1, acc);
      s[t] = acc;
    }

    // ---- online softmax; row reductions via v_permlane16 butterflies ----
    float corr[8];
#pragma unroll
    for (int r = 0; r < 8; ++r) {
      float mx = fmaxf(fmaxf(s[0][r], s[1][r]), fmaxf(s[2][r], s[3][r])) * scale;
      mx = row16_max(mx);
      const float nm = fmaxf(mrow[r], mx);
      corr[r] = __expf(mrow[r] - nm);
      mrow[r] = nm;
      float rs = 0.f;
#pragma unroll
      for (int t = 0; t < 4; ++t) {
        const float p = __expf(s[t][r] * scale - nm);
        s[t][r] = p;
        rs += p;
      }
      rs = row16_sum(rs);
      lrow[r] = lrow[r] * corr[r] + rs;
    }
#pragma unroll
    for (int t = 0; t < 4; ++t)
#pragma unroll
      for (int r = 0; r < 8; ++r) o[t][r] *= corr[r];

    // ---- transpose P (C-layout) -> LDS row-major [16 q][64 kk] ----
#pragma unroll
    for (int t = 0; t < 4; ++t)
#pragma unroll
      for (int r = 0; r < 8; ++r)
        pw[(r + 8 * half) * 64 + t * 16 + ln16] = (_Float16)s[t][r];
    // DS ops are in-order within a wave; staging slice is wave-private.

    // ---- O += P V  (K dim = 64 keys -> two K-steps of 32) ----
#pragma unroll
    for (int ks = 0; ks < 2; ++ks) {
      V16 ap;
      const _Float16* prow = pw + (size_t)ln16 * 64 + ks * 32;
      ap.h[0] = *(const v8h*)(prow + half * 16);
      ap.h[1] = *(const v8h*)(prow + half * 16 + 8);
#pragma unroll
      for (int t = 0; t < 4; ++t) {
        const _Float16* vp = Vb + (t * 16 + ln16) * 64 + ks * 32 + half * 16;
        v16h bv = *(const v16h*)vp;
        o[t] = WMMA_F16(ap.v, bv, o[t]);
      }
    }
    __syncthreads();  // protect current buffers before next TDM overwrite
  }

  // ---- finalize: divide by l, stage [128 q][64 d] tile, one TDM store ----
#pragma unroll
  for (int t = 0; t < 4; ++t) {
#pragma unroll
    for (int r = 0; r < 8; ++r)
      pw[(r + 8 * half) * 64 + t * 16 + ln16] = (_Float16)(o[t][r] / lrow[r]);
  }
  __syncthreads();
  if (wave == 0) {
    tdm_store((unsigned)(uintptr_t)&pbuf[0][0],
              AO + ((size_t)b * 2048 + q0) * 1024 + h * 64, 1,
              1024, 8192, 1024, 64, 128);
    __builtin_amdgcn_s_wait_tensorcnt(0);  // keep LDS alive until DMA done
  }
}

// ---------------------------------------------------------------------------
// Launcher
// ---------------------------------------------------------------------------
extern "C" void kernel_launch(void* const* d_in, const int* in_sizes, int n_in,
                              void* d_out, int out_size, void* d_ws, size_t ws_size,
                              hipStream_t stream) {
  const float* x  = (const float*)d_in[0];
  const float* wq = (const float*)d_in[1];
  const float* wk = (const float*)d_in[2];
  const float* wv = (const float*)d_in[3];
  const float* wo = (const float*)d_in[4];

  char* ws = (char*)d_ws;
  const size_t XB = (size_t)8192 * 1024 * 2;   // 16 MiB (f16 activations)
  const size_t WB = (size_t)1024 * 1024 * 2;   // 2 MiB per weight
  _Float16* xb  = (_Float16*)(ws);
  _Float16* wqb = (_Float16*)(ws + XB);
  _Float16* wkb = (_Float16*)(ws + XB + WB);
  _Float16* wvb = (_Float16*)(ws + XB + 2 * WB);
  _Float16* wob = (_Float16*)(ws + XB + 3 * WB);
  _Float16* qb  = (_Float16*)(ws + XB + 4 * WB);
  _Float16* kb  = (_Float16*)(ws + 2 * XB + 4 * WB);
  _Float16* vtb = (_Float16*)(ws + 3 * XB + 4 * WB);
  _Float16* aob = (_Float16*)(ws + 4 * XB + 4 * WB);

  // fp32 -> f16
  cvt_f32_f16_kernel<<<4096, 256, 0, stream>>>(x, xb, 8192 * 1024);
  cvt_f32_f16_kernel<<<1024, 256, 0, stream>>>(wq, wqb, 1024 * 1024);
  cvt_f32_f16_kernel<<<1024, 256, 0, stream>>>(wk, wkb, 1024 * 1024);
  cvt_f32_f16_kernel<<<1024, 256, 0, stream>>>(wv, wvb, 1024 * 1024);
  cvt_f32_f16_kernel<<<1024, 256, 0, stream>>>(wo, wob, 1024 * 1024);

  dim3 pgrid(16, 64);  // N/64, M/128
  proj_kernel<0><<<pgrid, 256, 0, stream>>>(xb, wqb, qb, nullptr);
  proj_kernel<1><<<pgrid, 256, 0, stream>>>(xb, wkb, kb, nullptr);
  proj_kernel<2><<<pgrid, 256, 0, stream>>>(xb, wvb, vtb, nullptr);

  dim3 agrid(64, 16);  // bh, qtiles
  attn_kernel<<<agrid, 256, 0, stream>>>(qb, kb, vtb, aob);

  proj_kernel<3><<<pgrid, 256, 0, stream>>>(aob, wob, nullptr, (float*)d_out);
}
